// MaskedTransformerBlock_69011534512667
// MI455X (gfx1250) — compile-verified, matrix operand order checked by reading, not run on hardware
//
#include <hip/hip_runtime.h>
#include <math.h>
#include <stdint.h>

// ---------------------------------------------------------------------------
// MI455X (gfx1250) transformer block.  Heavy math on v_wmma_f32_16x16x32_bf16
// (wave32, f32 accumulate); GEMM B-operand staged to LDS with the Tensor Data
// Mover (tensor_load_to_lds + s_wait_tensorcnt), double buffered.
// ---------------------------------------------------------------------------

typedef __bf16 bf16;
typedef __attribute__((ext_vector_type(16))) __bf16 v16bf;
typedef __attribute__((ext_vector_type(8)))  __bf16 v8bf;
typedef __attribute__((ext_vector_type(8)))  float  v8f;
typedef unsigned int u32x4 __attribute__((ext_vector_type(4)));
typedef int          i32x4 __attribute__((ext_vector_type(4)));
typedef int          i32x8 __attribute__((ext_vector_type(8)));

#define B_    2
#define S_    2048
#define D_    1024
#define H_    16
#define HD_   64
#define HID_  2752
#define SIXD_ 6144

#ifndef __has_builtin
#define __has_builtin(x) 0
#endif
#if defined(__HIP_DEVICE_COMPILE__) && __has_builtin(__builtin_amdgcn_tensor_load_to_lds)
#define HAVE_TDM 1
#else
#define HAVE_TDM 0
#endif

#if HAVE_TDM
#define TDM_WAIT_1() __builtin_amdgcn_s_wait_tensorcnt(1)
#define TDM_WAIT_0() __builtin_amdgcn_s_wait_tensorcnt(0)
#else
#define TDM_WAIT_1()
#define TDM_WAIT_0()
#endif

// ---------------------------------------------------------------------------
// Fragment helpers (CDNA5 WMMA VGPR layouts, CDNA5 ISA §7.12.2).
// A (16x32 bf16) and B-as-transposed (N x K row-major) share the pattern:
// lane L: two contiguous 16-byte chunks at k0 + (L>>4)*8 and k0+16+(L>>4)*8.
// ---------------------------------------------------------------------------
__device__ __forceinline__ v16bf frag_ld(const bf16* __restrict__ base,
                                         int row0, int k0, int ld) {
  const int lane = threadIdx.x & 31;
  const bf16* p = base + (size_t)(row0 + (lane & 15)) * ld
                       + (k0 + ((lane >> 4) << 3));
  v16bf f;
  *(v8bf*)&f         = *(const v8bf*)p;          // K offsets +0..7
  *(((v8bf*)&f) + 1) = *(const v8bf*)(p + 16);   // K offsets +16..23
  return f;
}

__device__ __forceinline__ v8f wmma_bf16(v16bf a, v16bf b, v8f c) {
  return __builtin_amdgcn_wmma_f32_16x16x32_bf16(false, a, false, b,
                                                 (short)0, c, false, false);
}

// ---------------------------------------------------------------------------
// TDM: DMA one [tileN rows x tileK cols] bf16 tile of BT[Ndim,Kdim] (row-major,
// K contiguous) into LDS.  D# per CDNA5 ISA:
//   group0: {count=1 | lds_addr | global_addr(57b) | type=2}
//   group1: {data_size=2B, tensor_dim0=K, tensor_dim1=N, tile_dim0=tileK,
//            tile_dim1=tileN, tensor_dim0_stride=K}
// Issued per-wave (EXEC ignored) -> call from one wave only; completion via
// TENSORcnt, publish to other waves with a workgroup barrier.
// This toolchain exposes the 6-arg builtin:
//   (u32x4 g0, i32x8 g1, i32x4 g2, i32x4 g3, i32x8 extra, i32 cpol)
// ---------------------------------------------------------------------------
__device__ __forceinline__ void tdm_load_b_tile(const bf16* gsrc, bf16* lds_dst,
                                                int Kdim, int Ndim,
                                                int tileK, int tileN) {
#if HAVE_TDM
  const uint64_t ga = (uint64_t)(uintptr_t)gsrc;        // byte address
  const uint32_t la = (uint32_t)(uintptr_t)lds_dst;     // LDS byte offset
  u32x4 g0 = { 1u,                                      // count=1 (valid D#)
               la,
               (uint32_t)ga,
               (uint32_t)((ga >> 32) & 0x01FFFFFFu) | (2u << 30) };  // type=2
  i32x8 g1;
  g1[0] = (int)0x00010000u;                             // data_size = 2 bytes
  g1[1] = (int)(((uint32_t)Kdim & 0xFFFFu) << 16);      // tensor_dim0 lo16
  g1[2] = (int)((((uint32_t)Kdim >> 16) & 0xFFFFu) |
                (((uint32_t)Ndim & 0xFFFFu) << 16));    // td0 hi | td1 lo
  g1[3] = (int)((((uint32_t)Ndim >> 16) & 0xFFFFu) |
                (((uint32_t)tileK & 0xFFFFu) << 16));   // td1 hi | tile_dim0
  g1[4] = (int)((uint32_t)tileN & 0xFFFFu);             // tile_dim1 (tile_dim2=0)
  g1[5] = (int)(uint32_t)Kdim;                          // tensor_dim0_stride lo32
  g1[6] = 0;                                            // stride0 hi | stride1 lo
  g1[7] = 0;
  i32x4 gz = { 0, 0, 0, 0 };                            // groups 2/3 unused (2D)
  i32x8 z8 = { 0, 0, 0, 0, 0, 0, 0, 0 };
  __builtin_amdgcn_tensor_load_to_lds(g0, g1, gz, gz, z8, 0);
#else
  // Fallback: cooperative copy by the calling wave (32 lanes), 16B chunks.
  const int lane = threadIdx.x & 31;
  const int chunksPerRow = tileK >> 3;
  for (int c = lane; c < tileN * chunksPerRow; c += 32) {
    const int r  = c / chunksPerRow;
    const int kk = (c % chunksPerRow) << 3;
    *(v8bf*)(lds_dst + r * tileK + kk) =
        *(const v8bf*)(gsrc + (size_t)r * Kdim + kk);
  }
#endif
}

// ---------------------------------------------------------------------------
// GEMM: C[M,N] (f32) = A[M,K] (bf16 row-major) x BT[N,K] (bf16 W^T row-major).
// Block = 128 threads = 4 waves (2x2); block tile 128x64, wave tile 64x32.
// B tile (64 x 64k, 8KB) TDM-staged to LDS, double buffered; A direct.
// M%128==0, N%64==0, K%64==0.
// ---------------------------------------------------------------------------
__global__ __launch_bounds__(128) void gemm_bf16_f32(
    const bf16* __restrict__ A, const bf16* __restrict__ BT,
    float* __restrict__ C, int M, int N, int K) {
  __shared__ __align__(16) bf16 smB[2][64 * 64];
  const int wid  = threadIdx.x >> 5;
  const int lane = threadIdx.x & 31;
  const int row0 = blockIdx.y * 128 + (wid >> 1) * 64;
  const int colw = (wid & 1) * 32;              // within 64-col block tile
  const int col0 = blockIdx.x * 64;
  const int nk   = K >> 6;                      // number of 64-wide k tiles
  v8f acc[4][2] = {};

  if (wid == 0)
    tdm_load_b_tile(BT + (size_t)col0 * K, &smB[0][0], K, N, 64, 64);

  for (int i = 0; i < nk; ++i) {
    const int kg = i << 6;
    if (wid == 0) {
      if (i + 1 < nk) {
        tdm_load_b_tile(BT + (size_t)col0 * K + (kg + 64),
                        &smB[(i + 1) & 1][0], K, N, 64, 64);
        TDM_WAIT_1();                           // tile i complete (in-order)
      } else {
        TDM_WAIT_0();
      }
    }
    __syncthreads();                            // publish buf[i&1]
    const bf16* bbuf = &smB[i & 1][0];
#pragma unroll
    for (int ks = 0; ks < 64; ks += 32) {
      v16bf a[4], b[2];
#pragma unroll
      for (int t = 0; t < 4; ++t) a[t] = frag_ld(A, row0 + 16 * t, kg + ks, K);
#pragma unroll
      for (int j = 0; j < 2; ++j) b[j] = frag_ld(bbuf, colw + 16 * j, ks, 64);
#pragma unroll
      for (int t = 0; t < 4; ++t)
#pragma unroll
        for (int j = 0; j < 2; ++j)
          acc[t][j] = wmma_bf16(a[t], b[j], acc[t][j]);
    }
    __syncthreads();                            // reads done before overwrite
  }

  const int n  = lane & 15;
  const int g8 = (lane >> 4) << 3;
#pragma unroll
  for (int t = 0; t < 4; ++t)
#pragma unroll
    for (int j = 0; j < 2; ++j) {
      float* cp = C + (size_t)(row0 + 16 * t + g8) * N + (col0 + colw + 16 * j + n);
#pragma unroll
      for (int r = 0; r < 8; ++r) cp[(size_t)r * N] = acc[t][j][r];
    }
}

// ---------------------------------------------------------------------------
// Fused SwiGLU FFN GEMM: OUT[M,N] (bf16) = silu(A@W1) * (A@W3)
// Wave tile 64x16, block tile 128x32 (two accumulator sets in VGPRs).
// ---------------------------------------------------------------------------
__global__ __launch_bounds__(128) void gemm_ffn_swiglu(
    const bf16* __restrict__ A, const bf16* __restrict__ B1T,
    const bf16* __restrict__ B3T, bf16* __restrict__ OUT,
    int M, int N, int K) {
  const int wid  = threadIdx.x >> 5;
  const int lane = threadIdx.x & 31;
  const int row0 = blockIdx.y * 128 + (wid >> 1) * 64;
  const int col0 = blockIdx.x * 32  + (wid & 1) * 16;
  v8f acc1[4] = {}, acc3[4] = {};
  for (int k0 = 0; k0 < K; k0 += 32) {
    v16bf b1 = frag_ld(B1T, col0, k0, K);
    v16bf b3 = frag_ld(B3T, col0, k0, K);
#pragma unroll
    for (int i = 0; i < 4; ++i) {
      v16bf a = frag_ld(A, row0 + 16 * i, k0, K);
      acc1[i] = wmma_bf16(a, b1, acc1[i]);
      acc3[i] = wmma_bf16(a, b3, acc3[i]);
    }
  }
  const int n  = lane & 15;
  const int g8 = (lane >> 4) << 3;
#pragma unroll
  for (int i = 0; i < 4; ++i)
#pragma unroll
    for (int r = 0; r < 8; ++r) {
      float u  = acc1[i][r];
      float t  = acc3[i][r];
      float sv = u / (1.f + expf(-u));
      OUT[(size_t)(row0 + 16 * i + g8 + r) * N + col0 + n] = (bf16)(sv * t);
    }
}

// ---------------------------------------------------------------------------
// Flash attention.  q,k bf16 [B,S,H,HD] (1/sqrt(HD) folded into q),
// vT bf16 [B,H,HD,S], mask f32 [S,S], out bf16 [B,S,H,HD].
// Block = 4 waves; wave owns 16 q-rows; 32-key steps; online softmax.
// P tile goes through LDS to convert C-layout -> A-fragment layout.
// ---------------------------------------------------------------------------
__global__ __launch_bounds__(128) void flash_attn(
    const bf16* __restrict__ q, const bf16* __restrict__ k,
    const bf16* __restrict__ vT, const float* __restrict__ mask,
    bf16* __restrict__ o) {
  __shared__ __align__(16) bf16 pbuf[4][16][32];
  const int wid  = threadIdx.x >> 5;
  const int lane = threadIdx.x & 31;
  const int bh = blockIdx.y;
  const int b  = bh >> 4;
  const int h  = bh & 15;
  const int q0 = blockIdx.x * 64 + wid * 16;
  const size_t base_qk = (size_t)b * ((size_t)S_ * H_ * HD_) + (size_t)h * HD_;
  const bf16* qb = q + base_qk;
  const bf16* kb = k + base_qk;
  const bf16* vb = vT + (size_t)bh * HD_ * S_;

  v16bf qf0 = frag_ld(qb, q0, 0,  H_ * HD_);
  v16bf qf1 = frag_ld(qb, q0, 32, H_ * HD_);

  v8f o_acc[4] = {};
  float mrow[8], lrow[8];
#pragma unroll
  for (int j = 0; j < 8; ++j) { mrow[j] = -1e30f; lrow[j] = 0.f; }

  const int n = lane & 15;
  const int g = lane >> 4;
  const float L2E = 1.44269504088896341f;

  for (int kb0 = 0; kb0 < S_; kb0 += 32) {
    if (kb0 + 32 < S_) {   // prefetch next K rows / V columns into caches
      __builtin_prefetch(kb + (size_t)(kb0 + 32 + lane) * (H_ * HD_), 0, 3);
      __builtin_prefetch(vb + (size_t)(lane & 3) * 16 * S_ + kb0 + 32, 0, 3);
    }
    v8f sc[2];
#pragma unroll
    for (int gk = 0; gk < 2; ++gk) {
      v8f s = {};
      v16bf kf0 = frag_ld(kb, kb0 + 16 * gk, 0,  H_ * HD_);
      v16bf kf1 = frag_ld(kb, kb0 + 16 * gk, 32, H_ * HD_);
      s = wmma_bf16(qf0, kf0, s);
      s = wmma_bf16(qf1, kf1, s);
      sc[gk] = s;
    }
    asm volatile("" ::: "memory");  // keep P stores below prior P loads (WAR)
#pragma unroll
    for (int j = 0; j < 8; ++j) {
      const int qr = q0 + j + g * 8;
      float s0 = sc[0][j] + mask[(size_t)qr * S_ + kb0 + n];
      float s1 = sc[1][j] + mask[(size_t)qr * S_ + kb0 + 16 + n];
      float mx = fmaxf(s0, s1);
#pragma unroll
      for (int off = 1; off < 16; off <<= 1)
        mx = fmaxf(mx, __shfl_xor(mx, off, 32));
      const float newm  = fmaxf(mrow[j], mx);
      const float scale = exp2f((mrow[j] - newm) * L2E);
      mrow[j] = newm;
      const float p0 = exp2f((s0 - newm) * L2E);
      const float p1 = exp2f((s1 - newm) * L2E);
      float rs = p0 + p1;
#pragma unroll
      for (int off = 1; off < 16; off <<= 1)
        rs += __shfl_xor(rs, off, 32);
      lrow[j] = lrow[j] * scale + rs;
#pragma unroll
      for (int t = 0; t < 4; ++t) o_acc[t][j] *= scale;
      pbuf[wid][j + g * 8][n]      = (bf16)p0;
      pbuf[wid][j + g * 8][16 + n] = (bf16)p1;
    }
    asm volatile("s_wait_dscnt 0" ::: "memory");  // P visible before re-read
    v16bf pf = frag_ld((const bf16*)&pbuf[wid][0][0], 0, 0, 32);
#pragma unroll
    for (int t = 0; t < 4; ++t) {
      v16bf vf = frag_ld(vb, 16 * t, kb0, S_);
      o_acc[t] = wmma_bf16(pf, vf, o_acc[t]);
    }
  }
#pragma unroll
  for (int t = 0; t < 4; ++t)
#pragma unroll
    for (int j = 0; j < 8; ++j) {
      const int qr = q0 + j + g * 8;
      o[base_qk + (size_t)qr * (H_ * HD_) + 16 * t + n] =
          (bf16)(o_acc[t][j] / lrow[j]);
    }
}

// ---------------------------------------------------------------------------
// Small helper kernels
// ---------------------------------------------------------------------------

// mod[b, n] = b_adaln[n] + sum_k silu(cond[b,k]) * w_adaln[k,n]   (2x6144)
__global__ __launch_bounds__(256) void adaln_mod(
    const float* __restrict__ cond, const float* __restrict__ w,
    const float* __restrict__ bias, float* __restrict__ mod) {
  const int o  = blockIdx.x * 256 + threadIdx.x;
  const int b  = o / SIXD_;
  const int nn = o % SIXD_;
  const float* cp = cond + (size_t)b * D_;
  float acc = bias[nn];
  for (int kk = 0; kk < D_; ++kk) {
    const float c = cp[kk];
    acc += (c / (1.f + expf(-c))) * w[(size_t)kk * SIXD_ + nn];
  }
  mod[o] = acc;
}

// LayerNorm over D then y = ln*(1+mod[sc]) + mod[sh], output bf16.
__global__ __launch_bounds__(256) void ln_mod(
    const float* __restrict__ x, const float* __restrict__ mod,
    bf16* __restrict__ out, int shOff, int scOff) {
  const int t = blockIdx.x;       // token index in [0, B*S)
  const int b = t >> 11;          // / S_
  const float* xp = x + (size_t)t * D_;
  __shared__ float s1[8], s2[8];
  float a = 0.f, sq = 0.f;
  for (int j = threadIdx.x; j < D_; j += 256) {
    const float v = xp[j]; a += v; sq += v * v;
  }
#pragma unroll
  for (int off = 16; off; off >>= 1) {
    a  += __shfl_xor(a, off, 32);
    sq += __shfl_xor(sq, off, 32);
  }
  if ((threadIdx.x & 31) == 0) { s1[threadIdx.x >> 5] = a; s2[threadIdx.x >> 5] = sq; }
  __syncthreads();
  float ta = 0.f, ts = 0.f;
#pragma unroll
  for (int w = 0; w < 8; ++w) { ta += s1[w]; ts += s2[w]; }
  const float mean = ta * (1.f / D_);
  const float var  = ts * (1.f / D_) - mean * mean;
  const float rstd = rsqrtf(var + 1e-5f);
  const float* mb = mod + (size_t)b * SIXD_;
  for (int j = threadIdx.x; j < D_; j += 256) {
    const float v = (xp[j] - mean) * rstd * (1.f + mb[scOff + j]) + mb[shOff + j];
    out[(size_t)t * D_ + j] = (bf16)v;
  }
}

// W^T + f32->bf16: wT[n,k] = (bf16) w[k,n].  grid (N/32, K/32), block (32,8).
__global__ void wtrans(const float* __restrict__ w, bf16* __restrict__ wT,
                       int K, int N) {
  __shared__ float tile[32][33];
  const int k0 = blockIdx.y * 32, n0 = blockIdx.x * 32;
  const int tx = threadIdx.x, ty = threadIdx.y;
  for (int r = ty; r < 32; r += 8)
    tile[r][tx] = w[(size_t)(k0 + r) * N + n0 + tx];
  __syncthreads();
  for (int r = ty; r < 32; r += 8)
    wT[(size_t)(n0 + r) * K + k0 + tx] = (bf16)tile[tx][r];
}

// RoPE + bf16 convert (+ optional 1/sqrt(HD) scale folded in for Q).
__global__ __launch_bounds__(256) void rope_cvt(
    const float* __restrict__ X, const float* __restrict__ cosp,
    const float* __restrict__ sinp, bf16* __restrict__ out, float scale) {
  const int i = blockIdx.x * 256 + threadIdx.x;   // pair index, B*S*H*HD/2
  const int token = i >> 9;                       // 512 pairs / token
  const int p  = i & 511;
  const int hh = p >> 5;
  const int j  = p & 31;
  const int s  = token & (S_ - 1);
  const float c  = cosp[s * 32 + j];
  const float sn = sinp[s * 32 + j];
  const size_t bi = (size_t)token * D_ + hh * HD_ + 2 * j;
  const float xr = X[bi], xi = X[bi + 1];
  out[bi]     = (bf16)((xr * c - xi * sn) * scale);
  out[bi + 1] = (bf16)((xr * sn + xi * c) * scale);
}

// V [B,S,H,HD] f32 -> Vt [B,H,HD,S] bf16.  grid (S/32, HD/32, B*H), block (32,8).
__global__ void vtrans(const float* __restrict__ Vf, bf16* __restrict__ vT) {
  __shared__ float tile[32][33];
  const int bh = blockIdx.z;
  const int b  = bh >> 4, h = bh & 15;
  const int s0 = blockIdx.x * 32, d0 = blockIdx.y * 32;
  const int tx = threadIdx.x, ty = threadIdx.y;
  for (int r = ty; r < 32; r += 8)
    tile[r][tx] = Vf[((size_t)(b * S_ + s0 + r) * H_ + h) * HD_ + d0 + tx];
  __syncthreads();
  for (int r = ty; r < 32; r += 8)
    vT[((size_t)bh * HD_ + d0 + r) * S_ + s0 + tx] = (bf16)tile[tx][r];
}

// out = x + mod[b, gateOff + j] * y      (f32 elementwise over [B*S, D])
__global__ __launch_bounds__(256) void resid_gate(
    const float* __restrict__ x, const float* __restrict__ y,
    const float* __restrict__ mod, int gateOff, float* __restrict__ out) {
  const size_t i = (size_t)blockIdx.x * 256 + threadIdx.x;
  const int j = (int)(i & (D_ - 1));
  const int b = (int)(i >> 21);                  // / (S_*D_)
  out[i] = x[i] + mod[(size_t)b * SIXD_ + gateOff + j] * y[i];
}

// ---------------------------------------------------------------------------
// Workspace layout (bytes).  Peak ~118 MB; lifetimes aliased as commented.
// ---------------------------------------------------------------------------
constexpr size_t SZ_WT_D = (size_t)D_ * D_ * 2;
constexpr size_t SZ_W1T  = (size_t)HID_ * D_ * 2;
constexpr size_t SZ_MOD  = (size_t)B_ * SIXD_ * 4;
constexpr size_t SZ_ABF  = (size_t)B_ * S_ * D_ * 2;
constexpr size_t SZ_AF   = (size_t)B_ * S_ * D_ * 4;

constexpr size_t O_WQT = 0;
constexpr size_t O_WKT = O_WQT + SZ_WT_D;
constexpr size_t O_WVT = O_WKT + SZ_WT_D;
constexpr size_t O_WOT = O_WVT + SZ_WT_D;
constexpr size_t O_W1T = O_WOT + SZ_WT_D;
constexpr size_t O_W3T = O_W1T + SZ_W1T;
constexpr size_t O_W2T = O_W3T + SZ_W1T;
constexpr size_t O_MOD = O_W2T + SZ_W1T;
constexpr size_t O_AQ  = (O_MOD + SZ_MOD + 255) & ~(size_t)255;
constexpr size_t O_AKV = O_AQ  + SZ_ABF;
constexpr size_t O_QF  = O_AKV + SZ_ABF;
constexpr size_t O_KF  = O_QF  + SZ_AF;
constexpr size_t O_VF  = O_KF  + SZ_AF;
constexpr size_t O_VT  = O_VF  + SZ_AF;
constexpr size_t O_FF  = O_VT  + SZ_ABF;
// aliases (disjoint lifetimes):
constexpr size_t O_QBF   = O_AQ;   // after GEMM1-3, aq dead
constexpr size_t O_KBF   = O_AKV;
constexpr size_t O_ATT   = O_QF;   // Qf dead after rope
constexpr size_t O_APROJ = O_KF;   // Kf dead after rope
constexpr size_t O_H     = O_VF;   // Vf dead after vtrans
constexpr size_t O_FX    = O_AQ;   // qbf dead after flash
constexpr size_t O_FFIN  = O_QF;   // 22.5MB spans Qf+part of Kf (both dead)

extern "C" void kernel_launch(void* const* d_in, const int* in_sizes, int n_in,
                              void* d_out, int out_size, void* d_ws, size_t ws_size,
                              hipStream_t stream) {
  const float* x_q   = (const float*)d_in[0];
  const float* x_kv  = (const float*)d_in[1];
  const float* fcos  = (const float*)d_in[2];
  const float* fsin  = (const float*)d_in[3];
  const float* amask = (const float*)d_in[4];
  const float* cond  = (const float*)d_in[5];
  const float* wq    = (const float*)d_in[6];
  const float* wk    = (const float*)d_in[7];
  const float* wv    = (const float*)d_in[8];
  const float* wo    = (const float*)d_in[9];
  const float* w_ada = (const float*)d_in[10];
  const float* b_ada = (const float*)d_in[11];
  const float* w1    = (const float*)d_in[12];
  const float* w2    = (const float*)d_in[13];
  const float* w3    = (const float*)d_in[14];
  float* out = (float*)d_out;
  char*  ws  = (char*)d_ws;

  float* mod    = (float*)(ws + O_MOD);
  bf16*  wqT    = (bf16*)(ws + O_WQT);
  bf16*  wkT    = (bf16*)(ws + O_WKT);
  bf16*  wvT    = (bf16*)(ws + O_WVT);
  bf16*  woT    = (bf16*)(ws + O_WOT);
  bf16*  w1T    = (bf16*)(ws + O_W1T);
  bf16*  w3T    = (bf16*)(ws + O_W3T);
  bf16*  w2T    = (bf16*)(ws + O_W2T);
  bf16*  aq     = (bf16*)(ws + O_AQ);
  bf16*  akv    = (bf16*)(ws + O_AKV);
  float* Qf     = (float*)(ws + O_QF);
  float* Kf     = (float*)(ws + O_KF);
  float* Vf     = (float*)(ws + O_VF);
  bf16*  qbf    = (bf16*)(ws + O_QBF);
  bf16*  kbf    = (bf16*)(ws + O_KBF);
  bf16*  vT     = (bf16*)(ws + O_VT);
  bf16*  att    = (bf16*)(ws + O_ATT);
  float* aproj  = (float*)(ws + O_APROJ);
  float* hbuf   = (float*)(ws + O_H);
  bf16*  fx     = (bf16*)(ws + O_FX);
  bf16*  ffin   = (bf16*)(ws + O_FFIN);
  float* ffout  = (float*)(ws + O_FF);

  const int NT = B_ * S_;                       // 4096 tokens
  const dim3 tb32x8(32, 8);

  // 1) adaLN modulation vector
  adaln_mod<<<(B_ * SIXD_) / 256, 256, 0, stream>>>(cond, w_ada, b_ada, mod);

  // 2) weight transposes -> bf16 W^T
  wtrans<<<dim3(D_ / 32, D_ / 32),   tb32x8, 0, stream>>>(wq, wqT, D_, D_);
  wtrans<<<dim3(D_ / 32, D_ / 32),   tb32x8, 0, stream>>>(wk, wkT, D_, D_);
  wtrans<<<dim3(D_ / 32, D_ / 32),   tb32x8, 0, stream>>>(wv, wvT, D_, D_);
  wtrans<<<dim3(D_ / 32, D_ / 32),   tb32x8, 0, stream>>>(wo, woT, D_, D_);
  wtrans<<<dim3(HID_ / 32, D_ / 32), tb32x8, 0, stream>>>(w1, w1T, D_, HID_);
  wtrans<<<dim3(HID_ / 32, D_ / 32), tb32x8, 0, stream>>>(w3, w3T, D_, HID_);
  wtrans<<<dim3(D_ / 32, HID_ / 32), tb32x8, 0, stream>>>(w2, w2T, HID_, D_);

  // 3) LN + adaLN(attn) -> bf16 activations
  ln_mod<<<NT, 256, 0, stream>>>(x_q,  mod, aq,  0, D_);
  ln_mod<<<NT, 256, 0, stream>>>(x_kv, mod, akv, 0, D_);

  // 4) Q/K/V projections (WMMA + TDM-staged B)
  gemm_bf16_f32<<<dim3(D_ / 64, NT / 128), 128, 0, stream>>>(aq,  wqT, Qf, NT, D_, D_);
  gemm_bf16_f32<<<dim3(D_ / 64, NT / 128), 128, 0, stream>>>(akv, wkT, Kf, NT, D_, D_);
  gemm_bf16_f32<<<dim3(D_ / 64, NT / 128), 128, 0, stream>>>(akv, wvT, Vf, NT, D_, D_);

  // 5) RoPE (+1/sqrt(HD) folded into Q), V transpose to [B,H,HD,S]
  const int npair = NT * (D_ / 2);
  rope_cvt<<<npair / 256, 256, 0, stream>>>(Qf, fcos, fsin, qbf, 0.125f);
  rope_cvt<<<npair / 256, 256, 0, stream>>>(Kf, fcos, fsin, kbf, 1.0f);
  vtrans<<<dim3(S_ / 32, HD_ / 32, B_ * H_), tb32x8, 0, stream>>>(Vf, vT);

  // 6) flash attention (WMMA scores + WMMA P.V)
  flash_attn<<<dim3(S_ / 64, B_ * H_), 128, 0, stream>>>(qbf, kbf, vT, amask, att);

  // 7) output projection, gated residual
  gemm_bf16_f32<<<dim3(D_ / 64, NT / 128), 128, 0, stream>>>(att, woT, aproj, NT, D_, D_);
  resid_gate<<<(NT * D_) / 256, 256, 0, stream>>>(x_q, aproj, mod, 2 * D_, hbuf);

  // 8) LN + adaLN(mlp), fused SwiGLU GEMM, down-proj, gated residual
  ln_mod<<<NT, 256, 0, stream>>>(hbuf, mod, fx, 3 * D_, 4 * D_);
  gemm_ffn_swiglu<<<dim3(HID_ / 32, NT / 128), 128, 0, stream>>>(fx, w1T, w3T, ffin,
                                                                 NT, HID_, D_);
  gemm_bf16_f32<<<dim3(D_ / 64, NT / 128), 128, 0, stream>>>(ffin, w2T, ffout,
                                                             NT, D_, HID_);
  resid_gate<<<(NT * D_) / 256, 256, 0, stream>>>(hbuf, ffout, mod, 5 * D_, out);
  (void)in_sizes; (void)n_in; (void)out_size; (void)ws_size;
}
